// CTCBeamSearchDecoder_54073638257108
// MI455X (gfx1250) — compile-verified, matrix operand order checked by reading, not run on hardware
//
#include <hip/hip_runtime.h>

#define BB 1024
#define TT 128
#define CC 63
#define VV 62   // non-blank classes
#define WW 8    // beam width
#define NEGF (-1.0e30f)
#define HB1 1000003u
#define HB2 2654435761u

typedef __attribute__((ext_vector_type(16))) _Float16 v16h;
typedef __attribute__((ext_vector_type(8)))  float    v8f;

__device__ __forceinline__ float lse2(float a, float b) {
  float m = fmaxf(a, b);
  float n = fminf(a, b);
  return m + __logf(1.0f + __expf(n - m));
}

// ---------------------------------------------------------------------------
// Kernel 1: per-frame log-sum-exp over the 63 classes for all B*T frames,
// computed as a reduction-by-matmul on the WMMA unit: D = exp(X) * ones.
// One wave handles 16 frames using V_WMMA_F32_16X16X32_F16 (two K=32 chunks,
// classes padded 63->64 with exp=0). Row sums land in the f32 accumulator.
// ---------------------------------------------------------------------------
__global__ __launch_bounds__(256) void ctc_lse_wmma(const float* __restrict__ logits,
                                                    float* __restrict__ denom) {
  const int tid     = blockIdx.x * blockDim.x + threadIdx.x;
  const int waveId  = tid >> 5;
  const int lane    = threadIdx.x & 31;
  const int frameBase = waveId * 16;          // 16 frames per wave, grid sized exactly
  const int r   = lane & 15;                  // matrix row (frame within group)
  const int grp = lane >> 4;                  // lane group selects K halves

  const float* row = logits + (size_t)(frameBase + r) * CC;

  v16h ones;
#pragma unroll
  for (int j = 0; j < 16; ++j) ones[j] = (_Float16)1.0f;

  v8f acc = {};
#pragma unroll
  for (int q = 0; q < 2; ++q) {               // K = 0..31, 32..63
    v16h a;
#pragma unroll
    for (int j = 0; j < 16; ++j) {
      // ISA 16-bit A layout (16x32): lanes 0-15 hold K {0..7,16..23},
      // lanes 16-31 hold K {8..15,24..31}; element j -> VGPR j/2 half j%2.
      int k = q * 32 + ((j < 8) ? (grp * 8 + j) : (16 + grp * 8 + (j - 8)));
      float v = (k < CC) ? __expf(row[k]) : 0.0f;   // pad class 63 with 0
      a[j] = (_Float16)v;
    }
    acc = __builtin_amdgcn_wmma_f32_16x16x32_f16(
        /*neg_a=*/false, a, /*neg_b=*/false, ones,
        /*c_mod=*/(short)0, acc, /*reuse_a=*/false, /*reuse_b=*/false);
  }

  // D layout: VGPR i, lanes 0-15 -> M=i ; lanes 16-31 -> M=8+i (all N equal).
  if (lane == 0 || lane == 16) {
    int mbase = frameBase + (lane >> 1);      // lane0 -> rows 0..7, lane16 -> rows 8..15
#pragma unroll
    for (int i = 0; i < 8; ++i) denom[mbase + i] = __logf(acc[i]);
  }
}

// ---------------------------------------------------------------------------
// Kernel 2: prefix beam search, one wave32 per batch element.
// ---------------------------------------------------------------------------
struct BeamStateView {
  const float* pb; const float* pnb; const int* plen; const int* last;
  const unsigned* h1; const unsigned* h2;
};

__device__ __forceinline__ void stay_cand(const BeamStateView& S, const float* lp, int wb,
                                          float& cpb, float& cpnb) {
  float p0 = S.pb[wb], pn0 = S.pnb[wb];
  cpb  = lse2(p0, pn0) + lp[CC - 1];          // emit blank
  cpnb = NEGF;
  int lw = S.last[wb];
  if (lw >= 0) {
    cpnb = pn0 + lp[lw];                      // repeat last char (collapsed)
    unsigned th1 = S.h1[wb], th2 = S.h2[wb];
    // merge extend(a, lw) whose resulting prefix hash equals this beam's hash
#pragma unroll
    for (int a = 0; a < WW; ++a) {
      if (S.h1[a] * HB1 + (unsigned)(lw + 1) == th1 &&
          S.h2[a] * HB2 + (unsigned)(lw + 1) == th2) {
        float base = (S.last[a] == lw) ? S.pb[a] : lse2(S.pb[a], S.pnb[a]);
        cpnb = lse2(cpnb, base + lp[lw]);
      }
    }
  }
}

__device__ __forceinline__ void ext_cand(const BeamStateView& S, const float* lp, int a, int c,
                                         float& cpb, float& cpnb) {
  cpb = NEGF;
  float base = (S.last[a] == c) ? S.pb[a] : lse2(S.pb[a], S.pnb[a]);
  cpnb = base + lp[c];
  // killed if some current beam's prefix == prefix_a + c (merged into its stay)
  unsigned eh1 = S.h1[a] * HB1 + (unsigned)(c + 1);
  unsigned eh2 = S.h2[a] * HB2 + (unsigned)(c + 1);
#pragma unroll
  for (int wb = 0; wb < WW; ++wb)
    if (S.last[wb] == c && S.h1[wb] == eh1 && S.h2[wb] == eh2) cpnb = NEGF;
}

__global__ __launch_bounds__(256) void ctc_beam(const float* __restrict__ logits,
                                                const float* denom,
                                                float* out) {
  __shared__ float    s_lp[8][64];
  __shared__ float    s_pb[2][8][WW],  s_pnb[2][8][WW];
  __shared__ int      s_plen[2][8][WW], s_last[2][8][WW];
  __shared__ unsigned s_h1[2][8][WW],  s_h2[2][8][WW];
  __shared__ short    s_pref[2][8][WW][TT];   // prefixes (labels 0..61)

  const int w    = threadIdx.x >> 5;          // wave within block
  const int lane = threadIdx.x & 31;
  const int b    = blockIdx.x * 8 + w;        // batch element

  if (lane < WW) {
    s_pb[0][w][lane]   = (lane == 0) ? 0.0f : NEGF;
    s_pnb[0][w][lane]  = NEGF;
    s_plen[0][w][lane] = 0;
    s_last[0][w][lane] = -1;
    s_h1[0][w][lane]   = 1u;
    s_h2[0][w][lane]   = 1u;
  }
  __syncthreads();

  const float* frame0 = logits + (size_t)b * TT * CC;

  for (int t = 0; t < TT; ++t) {
    // ---- stage normalized log-probs for this frame into LDS ----
    const float* fr = frame0 + t * CC;
    float dv = denom[b * TT + t];
    s_lp[w][lane] = fr[lane] - dv;
    int c2 = lane + 32;
    if (c2 < CC) s_lp[w][c2] = fr[c2] - dv;
    if (t + 1 < TT) __builtin_prefetch(fr + CC, 0, 3);   // global_prefetch_b8, next frame
    __syncthreads();

    const int ob = t & 1, nb = ob ^ 1;
    BeamStateView S = { s_pb[ob][w], s_pnb[ob][w], s_plen[ob][w],
                        s_last[ob][w], s_h1[ob][w], s_h2[ob][w] };
    const float* lp = s_lp[w];

    // ---- score 504 candidates (16 slots x 32 lanes, padded to 512) ----
    float sc[16];
#pragma unroll
    for (int s = 0; s < 16; ++s) {
      int n = s * 32 + lane;
      float cpb = NEGF, cpnb = NEGF;
      if (n < WW) {
        stay_cand(S, lp, n, cpb, cpnb);
      } else if (n < WW + WW * VV) {
        int e = n - WW;
        ext_cand(S, lp, e / VV, e % VV, cpb, cpnb);
      }
      sc[s] = lse2(cpb, cpnb);
    }

    // ---- iterative top-8: per-lane max, then wave argmax via ds_swizzle ----
    for (int i = 0; i < WW; ++i) {
      float best = -3.0e38f; int bslot = 0;
#pragma unroll
      for (int s = 0; s < 16; ++s)
        if (sc[s] > best) { best = sc[s]; bslot = s; }
      int blane = lane;
#pragma unroll
      for (int off = 16; off > 0; off >>= 1) {
        float obv = __shfl_xor(best, off, 32);
        int   osv = __shfl_xor(bslot, off, 32);
        int   olv = __shfl_xor(blane, off, 32);
        bool take = (obv > best) || (obv == best && olv < blane);
        if (take) { best = obv; bslot = osv; blane = olv; }
      }
#pragma unroll
      for (int s = 0; s < 16; ++s)              // kill winner (static indices)
        if (s == bslot && lane == blane) sc[s] = -3.0e38f;

      int nwin = bslot * 32 + blane;            // uniform across the wave

      // ---- recompute winner's full state ----
      int par, ch, nlast, nlen; unsigned nh1, nh2; float cpb, cpnb;
      if (nwin < WW) {
        par = nwin; ch = -1;
        stay_cand(S, lp, par, cpb, cpnb);
        nh1 = S.h1[par]; nh2 = S.h2[par];
        nlast = S.last[par]; nlen = S.plen[par];
      } else if (nwin < WW + WW * VV) {
        int e = nwin - WW; par = e / VV; int c = e % VV; ch = c;
        ext_cand(S, lp, par, c, cpb, cpnb);
        nh1 = S.h1[par] * HB1 + (unsigned)(c + 1);
        nh2 = S.h2[par] * HB2 + (unsigned)(c + 1);
        nlast = c; nlen = S.plen[par] + 1;
      } else {                                  // padding slot (never selected in practice)
        par = 0; ch = -1; cpb = NEGF; cpnb = NEGF;
        nh1 = 1u; nh2 = 1u; nlast = -1; nlen = 0;
      }
      bool alive = best > 0.5f * NEGF;
      if (!alive) { cpb = NEGF; cpnb = NEGF; }

      if (lane == 0) {
        s_pb[nb][w][i]   = cpb;  s_pnb[nb][w][i]  = cpnb;
        s_plen[nb][w][i] = nlen; s_last[nb][w][i] = nlast;
        s_h1[nb][w][i]   = nh1;  s_h2[nb][w][i]   = nh2;
      }
      // copy parent prefix (128 shorts, 4 per lane), then append
#pragma unroll
      for (int j = 0; j < 4; ++j)
        s_pref[nb][w][i][lane * 4 + j] = s_pref[ob][w][par][lane * 4 + j];
      if (ch >= 0 && lane == 0) s_pref[nb][w][i][nlen - 1] = (short)ch;
    }
    __syncthreads();
  }

  // ---- emit top path (beam 0); final state is in buffer 0 since TT is even ----
  const int fb = 0;
  int len0 = s_plen[fb][w][0];
#pragma unroll
  for (int j = 0; j < 4; ++j) {
    int pos = lane * 4 + j;
    out[(size_t)b * TT + pos] = (pos < len0) ? (float)s_pref[fb][w][0][pos] : -1.0f;
  }
  if (lane == 0) {
    out[BB * TT + b] = (float)len0;
    out[BB * TT + BB + b] = __expf(lse2(s_pb[fb][w][0], s_pnb[fb][w][0]));
  }
}

// ---------------------------------------------------------------------------
extern "C" void kernel_launch(void* const* d_in, const int* in_sizes, int n_in,
                              void* d_out, int out_size, void* d_ws, size_t ws_size,
                              hipStream_t stream) {
  (void)in_sizes; (void)n_in; (void)out_size;
  const float* logits = (const float*)d_in[0];
  float* out = (float*)d_out;
  // 512 KB of per-frame logsumexp denominators: prefer workspace, fall back to
  // the decoded region of d_out (each wave finishes reading its batch's denoms
  // before overwriting that exact region with decoded labels).
  float* denom = (ws_size >= (size_t)BB * TT * sizeof(float)) ? (float*)d_ws : out;

  // 131072 frames, 16 per wave, 8 waves per block -> 1024 blocks
  ctc_lse_wmma<<<(BB * TT) / (16 * 8), 256, 0, stream>>>(logits, denom);
  // 1024 batch elements, 1 wave each, 8 waves per block -> 128 blocks
  ctc_beam<<<BB / 8, 256, 0, stream>>>(logits, denom, out);
}
